// _GRUD_3281355014586
// MI455X (gfx1250) — compile-verified, matrix operand order checked by reading, not run on hardware
//
#include <hip/hip_runtime.h>

// GRU-D forward for MI455X (gfx1250): bf16 WMMA GEMMs + fused epilogues.
// B=1024, T=64, F=256, H=1024, C=2.
// v2: double-buffered LDS weight tiles (1 barrier per k-step) and 2x4 WMMA
//     tiles per wave (8 wmma per k-step, higher intensity per barrier).

#define B_ 1024
#define T_ 64
#define F_ 256
#define H_ 1024
#define C_ 2

typedef __attribute__((ext_vector_type(16))) __bf16 v16bf;
typedef __attribute__((ext_vector_type(8)))  float  v8f;
typedef __attribute__((ext_vector_type(4)))  unsigned int u32x4;

union FragBF { u32x4 u[2]; v16bf v; };

__device__ inline unsigned short f2bf(float f) {
  unsigned int u = __float_as_uint(f);
  u += 0x7FFFu + ((u >> 16) & 1u);   // round-to-nearest-even
  return (unsigned short)(u >> 16);
}

// ---------------------------------------------------------------------------
// Core WMMA GEMM accumulator (double-buffered LDS weight tiles).
// A: [M x K] row-major bf16 bits. W: [N x K] row-major bf16 bits.
// Block = 256 threads = 8 waves stacked in M. Macro tile: 256(M) x 64(N).
// Each wave owns 2x4 16x16 C tiles: rows m0..m0+31, cols n0b + {0,16,32,48}.
// The 64x32 B tile is staged in LDS (2 buffers, 8 KB) and reused by all waves;
// tile s+1 is written while tile s is consumed -> one barrier per k-step.
// bf16 A-fragment layout (ISA 7.12.2): lane m = lane&15; k-half = (lane<16)?0:8;
// elements 0..7 -> k = half+e, elements 8..15 -> k = half+16+(e-8).
// ---------------------------------------------------------------------------
__device__ inline void wmma_gemm_acc(const unsigned short* __restrict__ A, int lda,
                                     const unsigned short* __restrict__ W, int ldw,
                                     int K, int m0, int n0b,
                                     unsigned short* sB,   // 2 * 64 * 32 entries
                                     v8f acc[2][4]) {
  const int tid   = threadIdx.x;
  const int lane  = tid & 31;
  const int nlo   = lane & 15;
  const int khalf = (lane < 16) ? 0 : 8;
  const int ldr   = tid >> 2;        // 0..63 : n row of B tile
  const int ldq   = (tid & 3) * 8;   // 0,8,16,24 : k offset (8 bf16 = 16B)
  const unsigned short* wsrc = W + (size_t)(n0b + ldr) * (size_t)ldw + ldq;

  const int nsteps = K >> 5;

  // Preload tile 0 into buffer 0.
  *(u32x4*)(sB + ldr * 32 + ldq) = *(const u32x4*)(wsrc);
  __syncthreads();

  for (int s = 0; s < nsteps; ++s) {
    const int k0 = s << 5;
    unsigned short* cur = sB + ((s & 1) << 11);        // 64*32 = 2048 entries
    unsigned short* nxt = sB + (((s + 1) & 1) << 11);

    // Stage next weight tile while this one is consumed.
    if (s + 1 < nsteps) {
      u32x4 w = *(const u32x4*)(wsrc + k0 + 32);
      *(u32x4*)(nxt + ldr * 32 + ldq) = w;
    }

    // Two A fragments (rows m0.., m0+16..) straight from global (L2-resident).
    FragBF a0, a1;
    const unsigned short* ap0 = A + (size_t)(m0 + nlo) * (size_t)lda + k0 + khalf;
    const unsigned short* ap1 = ap0 + (size_t)16 * (size_t)lda;
    a0.u[0] = *(const u32x4*)(ap0);
    a0.u[1] = *(const u32x4*)(ap0 + 16);
    a1.u[0] = *(const u32x4*)(ap1);
    a1.u[1] = *(const u32x4*)(ap1 + 16);

#pragma unroll
    for (int nt = 0; nt < 4; ++nt) {
      FragBF b;
      const unsigned short* bp = cur + (nt * 16 + nlo) * 32 + khalf;
      b.u[0] = *(const u32x4*)(bp);
      b.u[1] = *(const u32x4*)(bp + 16);
      acc[0][nt] = __builtin_amdgcn_wmma_f32_16x16x32_bf16(
          false, a0.v, false, b.v, (short)0, acc[0][nt], false, false);
      acc[1][nt] = __builtin_amdgcn_wmma_f32_16x16x32_bf16(
          false, a1.v, false, b.v, (short)0, acc[1][nt], false, false);
    }
    __syncthreads();   // nxt fully staged AND all reads of cur complete
  }
}

// ---------------------------------------------------------------------------
// Prep kernels (run once, fully parallel)
// ---------------------------------------------------------------------------
__global__ void k_prep_wg(const float* __restrict__ w_gh, unsigned short* __restrict__ WG) {
  int i = blockIdx.x * blockDim.x + threadIdx.x;            // H_*F_
  if (i < H_ * F_) WG[i] = f2bf(w_gh[i]);
}

__global__ void k_prep_wxm(const float* __restrict__ w_ih, unsigned short* __restrict__ WXM) {
  int i = blockIdx.x * blockDim.x + threadIdx.x;            // 3H * 2F
  if (i >= 3 * H_ * 2 * F_) return;
  int j = i / (2 * F_);
  int k = i - j * (2 * F_);
  // cols 0..255: x_replaced part (w_ih col k); cols 256..511: mask part (w_ih col 1280+(k-256))
  int src = (k < F_) ? k : (F_ + H_ + (k - F_));
  WXM[i] = f2bf(w_ih[(size_t)j * (2 * F_ + H_) + src]);
}

__global__ void k_prep_wh6(const float* __restrict__ w_ih, const float* __restrict__ w_hh,
                           unsigned short* __restrict__ WH6) {
  size_t i = (size_t)blockIdx.x * blockDim.x + threadIdx.x; // 6H * H
  if (i >= (size_t)6 * H_ * H_) return;
  int j = (int)(i / H_);
  int k = (int)(i - (size_t)j * H_);
  float v = (j < 3 * H_) ? w_ih[(size_t)j * (2 * F_ + H_) + F_ + k]
                         : w_hh[(size_t)(j - 3 * H_) * H_ + k];
  WH6[i] = f2bf(v);
}

// gamma_x / x_replaced / bf16 casts for all timesteps (h-independent)
__global__ void k_prep_inputs(const float* __restrict__ X, const float* __restrict__ M,
                              const float* __restrict__ D, const float* __restrict__ MEAN,
                              const float* __restrict__ XL,
                              const float* __restrict__ w_gx, const float* __restrict__ b_gx,
                              unsigned short* __restrict__ XM,   // [T][B][2F]
                              unsigned short* __restrict__ DBF)  // [T][B][F]
{
  size_t i = (size_t)blockIdx.x * blockDim.x + threadIdx.x;  // B*T*F
  if (i >= (size_t)B_ * T_ * F_) return;
  int f = (int)(i % F_);
  int t = (int)((i / F_) % T_);
  int b = (int)(i / ((size_t)F_ * T_));
  float d  = D[i];
  float m  = M[i];
  float gx = expf(-fmaxf(d * w_gx[f] + b_gx[f], 0.0f));
  float xh = gx * XL[i] + (1.0f - gx) * MEAN[(size_t)b * F_ + f];
  float xr = m * X[i] + (1.0f - m) * xh;
  size_t row = (size_t)t * B_ + b;
  XM[row * (2 * F_) + f]      = f2bf(xr);
  XM[row * (2 * F_) + F_ + f] = f2bf(m);
  DBF[row * F_ + f]           = f2bf(d);
}

__global__ void k_zero(float* __restrict__ p, size_t n) {
  size_t i = (size_t)blockIdx.x * blockDim.x + threadIdx.x;
  if (i < n) p[i] = 0.0f;
}

// ---------------------------------------------------------------------------
// Per-step kernel 1: gamma_h GEMM + decay.
// acc = D_t @ w_gh^T ; gamma = exp(-relu(acc + b_gh)) ; hd = h * gamma
// ---------------------------------------------------------------------------
__global__ void k_gamma_hd(const unsigned short* __restrict__ DBFt, // [B x F]
                           const unsigned short* __restrict__ WG,   // [H x F]
                           const float* __restrict__ b_gh,
                           const float* __restrict__ h,             // [B x H]
                           float* __restrict__ hdf,                 // [B x H]
                           unsigned short* __restrict__ hdb)        // [B x H] bf16
{
  __shared__ unsigned short sB[2 * 64 * 32];
  const int wave = threadIdx.x >> 5;
  const int lane = threadIdx.x & 31;
  const int m0   = blockIdx.x * 256 + wave * 32;
  const int n0b  = blockIdx.y * 64;
  v8f zero = {};
  v8f acc[2][4] = {{zero, zero, zero, zero}, {zero, zero, zero, zero}};

  wmma_gemm_acc(DBFt, F_, WG, F_, F_, m0, n0b, sB, acc);

  const int nlo = lane & 15;
  const int mo  = (lane >> 4) * 8;
#pragma unroll
  for (int mt = 0; mt < 2; ++mt) {
#pragma unroll
    for (int nt = 0; nt < 4; ++nt) {
      int n = n0b + nt * 16 + nlo;
      float bg = b_gh[n];
#pragma unroll
      for (int r = 0; r < 8; ++r) {
        int m = m0 + mt * 16 + mo + r;
        float g = expf(-fmaxf(acc[mt][nt][r] + bg, 0.0f));
        size_t idx = (size_t)m * H_ + n;
        float hv = h[idx] * g;
        hdf[idx] = hv;
        hdb[idx] = f2bf(hv);
      }
    }
  }
}

// ---------------------------------------------------------------------------
// Per-step kernel 2: combined gate GEMM.
// OUT6[:, 0:3072]    = b_ih + HD @ w_ih_h^T + XM_t @ w_ih_xm^T   (gi)
// OUT6[:, 3072:6144] = b_hh + HD @ w_hh^T                        (gh)
// ---------------------------------------------------------------------------
__global__ void k_gates_gemm(const unsigned short* __restrict__ HDB, // [B x H]
                             const unsigned short* __restrict__ WH6, // [6H x H]
                             const unsigned short* __restrict__ XMt, // [B x 2F]
                             const unsigned short* __restrict__ WXM, // [3H x 2F]
                             const float* __restrict__ b_ih,
                             const float* __restrict__ b_hh,
                             float* __restrict__ OUT6)               // [B x 6H]
{
  __shared__ unsigned short sB[2 * 64 * 32];
  const int wave = threadIdx.x >> 5;
  const int lane = threadIdx.x & 31;
  const int m0   = blockIdx.x * 256 + wave * 32;
  const int n0b  = blockIdx.y * 64;
  v8f zero = {};
  v8f acc[2][4] = {{zero, zero, zero, zero}, {zero, zero, zero, zero}};

  wmma_gemm_acc(HDB, H_, WH6, H_, H_, m0, n0b, sB, acc);
  if (n0b < 3 * H_) {  // block-uniform: gi columns also get the x/mask contribution
    wmma_gemm_acc(XMt, 2 * F_, WXM, 2 * F_, 2 * F_, m0, n0b, sB, acc);
  }

  const int nlo = lane & 15;
  const int mo  = (lane >> 4) * 8;
#pragma unroll
  for (int mt = 0; mt < 2; ++mt) {
#pragma unroll
    for (int nt = 0; nt < 4; ++nt) {
      int n = n0b + nt * 16 + nlo;
      float bias = (n < 3 * H_) ? b_ih[n] : b_hh[n - 3 * H_];
#pragma unroll
      for (int r = 0; r < 8; ++r) {
        int m = m0 + mt * 16 + mo + r;
        OUT6[(size_t)m * (6 * H_) + n] = acc[mt][nt][r] + bias;
      }
    }
  }
}

// ---------------------------------------------------------------------------
// Per-step kernel 3: GRU gating (elementwise).
// ---------------------------------------------------------------------------
__global__ void k_gate_update(const float* __restrict__ OUT6,
                              const float* __restrict__ hdf,
                              float* __restrict__ h) {
  size_t i = (size_t)blockIdx.x * blockDim.x + threadIdx.x;  // B*H
  if (i >= (size_t)B_ * H_) return;
  int b = (int)(i >> 10);
  int c = (int)(i & (H_ - 1));
  const float* row = OUT6 + (size_t)b * (6 * H_);
  float ir = row[c],            iz = row[H_ + c],     in_ = row[2 * H_ + c];
  float hr = row[3 * H_ + c],   hz = row[4 * H_ + c], hn  = row[5 * H_ + c];
  float r = 1.0f / (1.0f + expf(-(ir + hr)));
  float z = 1.0f / (1.0f + expf(-(iz + hz)));
  float n = tanhf(in_ + r * hn);
  float hd = hdf[i];
  h[i] = (1.0f - z) * n + z * hd;
}

// ---------------------------------------------------------------------------
// Classifier + softmax (tiny tail: 1024 rows, K=1024, C=2).
// ---------------------------------------------------------------------------
__global__ void k_cls(const float* __restrict__ h, const float* __restrict__ w_cls,
                      const float* __restrict__ b_cls, float* __restrict__ out) {
  int row = blockIdx.x * blockDim.x + threadIdx.x;
  if (row >= B_) return;
  float a0 = b_cls[0], a1 = b_cls[1];
  const float* hr = h + (size_t)row * H_;
  for (int k = 0; k < H_; ++k) {
    float hv = hr[k];
    a0 += hv * w_cls[k];
    a1 += hv * w_cls[H_ + k];
  }
  float mx = fmaxf(a0, a1);
  float e0 = expf(a0 - mx), e1 = expf(a1 - mx);
  float s = e0 + e1;
  out[(size_t)row * C_ + 0] = e0 / s;
  out[(size_t)row * C_ + 1] = e1 / s;
}

// ---------------------------------------------------------------------------
// Host launch
// ---------------------------------------------------------------------------
extern "C" void kernel_launch(void* const* d_in, const int* in_sizes, int n_in,
                              void* d_out, int out_size, void* d_ws, size_t ws_size,
                              hipStream_t stream) {
  const float* X     = (const float*)d_in[0];
  const float* MM    = (const float*)d_in[1];
  const float* DL    = (const float*)d_in[2];
  const float* MEAN  = (const float*)d_in[3];
  const float* XL    = (const float*)d_in[4];
  const float* w_gh  = (const float*)d_in[5];
  const float* b_gh  = (const float*)d_in[6];
  const float* w_gx  = (const float*)d_in[7];
  const float* b_gx  = (const float*)d_in[8];
  const float* w_ih  = (const float*)d_in[9];
  const float* w_hh  = (const float*)d_in[10];
  const float* b_ih  = (const float*)d_in[11];
  const float* b_hh  = (const float*)d_in[12];
  const float* w_cls = (const float*)d_in[13];
  const float* b_cls = (const float*)d_in[14];
  float* out = (float*)d_out;

  // Carve workspace (256B aligned).
  char* p = (char*)d_ws;
  auto carve = [&](size_t bytes) -> void* {
    void* r = (void*)p;
    p += (bytes + 255) & ~(size_t)255;
    return r;
  };
  unsigned short* WG  = (unsigned short*)carve((size_t)H_ * F_ * 2);            // 0.5 MB
  unsigned short* WXM = (unsigned short*)carve((size_t)3 * H_ * 2 * F_ * 2);    // 3 MB
  unsigned short* WH6 = (unsigned short*)carve((size_t)6 * H_ * H_ * 2);        // 12 MB
  unsigned short* DBF = (unsigned short*)carve((size_t)T_ * B_ * F_ * 2);       // 32 MB
  unsigned short* XM  = (unsigned short*)carve((size_t)T_ * B_ * 2 * F_ * 2);   // 64 MB
  unsigned short* HDB = (unsigned short*)carve((size_t)B_ * H_ * 2);            // 2 MB
  float* Hs   = (float*)carve((size_t)B_ * H_ * 4);                             // 4 MB
  float* HDF  = (float*)carve((size_t)B_ * H_ * 4);                             // 4 MB
  float* OUT6 = (float*)carve((size_t)B_ * 6 * H_ * 4);                         // 24 MB

  const int TB = 256;

  // One-time preprocessing (all h-independent work hoisted out of recurrence).
  k_prep_wg <<<(H_ * F_ + TB - 1) / TB, TB, 0, stream>>>(w_gh, WG);
  k_prep_wxm<<<(3 * H_ * 2 * F_ + TB - 1) / TB, TB, 0, stream>>>(w_ih, WXM);
  {
    size_t n = (size_t)6 * H_ * H_;
    k_prep_wh6<<<(unsigned)((n + TB - 1) / TB), TB, 0, stream>>>(w_ih, w_hh, WH6);
  }
  {
    size_t n = (size_t)B_ * T_ * F_;
    k_prep_inputs<<<(unsigned)((n + TB - 1) / TB), TB, 0, stream>>>(
        X, MM, DL, MEAN, XL, w_gx, b_gx, XM, DBF);
  }
  {
    size_t n = (size_t)B_ * H_;
    k_zero<<<(unsigned)((n + TB - 1) / TB), TB, 0, stream>>>(Hs, n);
  }

  // Sequential recurrence: 3 kernels per timestep.
  dim3 g1(B_ / 256, H_ / 64);          // 4 x 16
  dim3 g2(B_ / 256, (6 * H_) / 64);    // 4 x 96
  unsigned g3 = (unsigned)(((size_t)B_ * H_ + TB - 1) / TB);
  for (int t = 0; t < T_; ++t) {
    const unsigned short* DBFt = DBF + (size_t)t * B_ * F_;
    const unsigned short* XMt  = XM  + (size_t)t * B_ * 2 * F_;
    k_gamma_hd  <<<g1, TB, 0, stream>>>(DBFt, WG, b_gh, Hs, HDF, HDB);
    k_gates_gemm<<<g2, TB, 0, stream>>>(HDB, WH6, XMt, WXM, b_ih, b_hh, OUT6);
    k_gate_update<<<g3, TB, 0, stream>>>(OUT6, HDF, Hs);
  }

  k_cls<<<(B_ + TB - 1) / TB, TB, 0, stream>>>(Hs, w_cls, b_cls, out);
}
